// AutoregressiveLSTMDecoder_56229711839903
// MI455X (gfx1250) — matrix operand order repaired and checked by reading, not compiled, not run on hardware
//
#include <hip/hip_runtime.h>
#include <hip/hip_bf16.h>

// ---------------------------------------------------------------------------
// Autoregressive LSTM decoder on gfx1250 (MI455X), wave32 + WMMA f16.
//   B=256, T_IN=512, E=512, H=512, 4H=2048, OUT=2, SEQ=512
// Per step: gates = x_t@W_ih^T + h@W_hh^T + bias (dual-GEMM, f16 in / f32
// accum via v_wmma_f32_16x16x32_f16), then cell update + FC head.
// Weights packed once per call into WMMA B-fragment order; x_t staged to f16
// by the previous step's cell kernel so the GEMM is pure-f16 fed.
// Register blocking: each wave owns 2 m-tiles x 2 n-tiles (A reused 2x,
// B reused 2x -> B-from-L2 traffic halved vs 1x4 blocking).
// ---------------------------------------------------------------------------

typedef __attribute__((ext_vector_type(16))) _Float16 v16h;
typedef __attribute__((ext_vector_type(8)))  _Float16 v8h;
typedef __attribute__((ext_vector_type(8)))  float    v8f;

#define LSTM_B   256
#define LSTM_T   512
#define LSTM_E   512
#define LSTM_H   512
#define LSTM_G   2048   // 4*H
#define LSTM_OUT 2

// ---------------------------------------------------------------------------
// Pack W [2048 x 512] (row-major f32; gates = A @ W^T so B[k][n] = W[n][k])
// into WMMA 16-bit B-fragment order:
//   packed[((nt*16 + kt)*32 + lane)*16 + j] = W[nt*16 + lane%16][kt*32 + (lane/16)*8 + (j%8) + (j/8)*16]
// so each lane's B fragment is one contiguous 32B chunk.
// ---------------------------------------------------------------------------
__global__ __launch_bounds__(256)
void lstm_pack_w(const float* __restrict__ W, _Float16* __restrict__ P) {
    int idx  = blockIdx.x * 256 + threadIdx.x;        // packed-order index
    int j    = idx & 15;
    int lane = (idx >> 4) & 31;
    int ktnt = idx >> 9;
    int kt   = ktnt & 15;
    int nt   = ktnt >> 4;
    int n    = nt * 16 + (lane & 15);
    int k    = kt * 32 + (lane >> 4) * 8 + (j & 7) + (j >> 3) * 16;
    P[idx] = (_Float16)W[(size_t)n * LSTM_H + k];
}

// Zero h/c state and stage x[:,0,:] to f16.
__global__ __launch_bounds__(256)
void lstm_init_state(float* __restrict__ c, _Float16* __restrict__ hf,
                     _Float16* __restrict__ xh, const float* __restrict__ x) {
    int i = blockIdx.x * 256 + threadIdx.x;           // 0 .. B*H-1
    int b = i >> 9;
    int j = i & 511;
    c[i]  = 0.0f;
    hf[i] = (_Float16)0.0f;
    xh[i] = (_Float16)x[((size_t)b * LSTM_T + 0) * LSTM_E + j];
}

// ---------------------------------------------------------------------------
// Per-step fused dual-GEMM: gates[B,4H] = xh @ W_ih^T + hf @ W_hh^T + bias.
// 512 waves: wave = (mg in [0,8)) x (ng in [0,64)); 2 m-tiles x 2 n-tiles.
// All 16 fragment loads per K-chunk are issued before the 8-WMMA chain so
// the scheduler can clause them and overlap with the matrix pipe.
// ---------------------------------------------------------------------------
__global__ __launch_bounds__(128)
void lstm_gemm_step(const _Float16* __restrict__ xh,   // [B, E] f16 (this step's x_t)
                    const _Float16* __restrict__ hf,   // [B, H] f16
                    const _Float16* __restrict__ WihP, // packed B-frags
                    const _Float16* __restrict__ WhhP, // packed B-frags
                    const float* __restrict__ b_ih,
                    const float* __restrict__ b_hh,
                    float* __restrict__ gates) {       // [B, 4H] f32
    const int gwave = (blockIdx.x * 128 + (int)threadIdx.x) >> 5;  // 0..511
    const int lane  = threadIdx.x & 31;
    const int mg    = gwave >> 6;        // 0..7  : pair of m-tiles
    const int ng    = gwave & 63;        // 0..63 : pair of n-tiles
    const int row   = lane & 15;         // A: M in tile; B/C/D: N in tile
    const int hi    = lane >> 4;         // A: k-half; B: k-group; C/D: M-half

    const _Float16* xrow0 = xh + (size_t)((mg * 2 + 0) * 16 + row) * LSTM_E;
    const _Float16* xrow1 = xh + (size_t)((mg * 2 + 1) * 16 + row) * LSTM_E;
    const _Float16* hrow0 = hf + (size_t)((mg * 2 + 0) * 16 + row) * LSTM_H;
    const _Float16* hrow1 = hf + (size_t)((mg * 2 + 1) * 16 + row) * LSTM_H;

    v8f acc[2][2];
    for (int i = 0; i < 2; ++i) {
        const int n = (ng * 2 + i) * 16 + row;
        const float bias = b_ih[n] + b_hh[n];
        for (int r = 0; r < 8; ++r) { acc[0][i][r] = bias; acc[1][i][r] = bias; }
    }

    for (int kt = 0; kt < 16; ++kt) {
        const int k0 = kt * 32 + hi * 8;

        // ---- all loads first (one clause, partial loadcnt waits) ----
        v16h bi[2], bh[2];
        for (int i = 0; i < 2; ++i) {
            const size_t pb = ((size_t)((ng * 2 + i) * 16 + kt) * 32 + lane) * 16;
            bi[i] = *(const v16h*)(WihP + pb);
            bh[i] = *(const v16h*)(WhhP + pb);
        }
        // A layout: halves 0..7 <-> K=k0..k0+7, halves 8..15 <-> K=k0+16..k0+23
        v8h xa0 = *(const v8h*)(xrow0 + k0);
        v8h xb0 = *(const v8h*)(xrow0 + k0 + 16);
        v8h xa1 = *(const v8h*)(xrow1 + k0);
        v8h xb1 = *(const v8h*)(xrow1 + k0 + 16);
        v8h ha0 = *(const v8h*)(hrow0 + k0);
        v8h hb0 = *(const v8h*)(hrow0 + k0 + 16);
        v8h ha1 = *(const v8h*)(hrow1 + k0);
        v8h hb1 = *(const v8h*)(hrow1 + k0 + 16);

        v16h ax[2], ah[2];
        for (int q = 0; q < 8; ++q) {
            ax[0][q] = xa0[q]; ax[0][8 + q] = xb0[q];
            ax[1][q] = xa1[q]; ax[1][8 + q] = xb1[q];
            ah[0][q] = ha0[q]; ah[0][8 + q] = hb0[q];
            ah[1][q] = ha1[q]; ah[1][8 + q] = hb1[q];
        }

        // ---- 8 WMMAs: 4 independent accumulator chains of depth 2 ----
        for (int mi = 0; mi < 2; ++mi) {
            for (int i = 0; i < 2; ++i) {
                acc[mi][i] = __builtin_amdgcn_wmma_f32_16x16x32_f16(
                    false, ax[mi], false, bi[i], (short)0, acc[mi][i], false, false);
                acc[mi][i] = __builtin_amdgcn_wmma_f32_16x16x32_f16(
                    false, ah[mi], false, bh[i], (short)0, acc[mi][i], false, false);
            }
        }
    }

    // C/D layout: lane -> N = row, VGPR r -> M = r + 8*hi.
    for (int mi = 0; mi < 2; ++mi) {
        for (int i = 0; i < 2; ++i) {
            const int n = (ng * 2 + i) * 16 + row;
            for (int r = 0; r < 8; ++r) {
                gates[(size_t)((mg * 2 + mi) * 16 + r + 8 * hi) * LSTM_G + n] =
                    acc[mi][i][r];
            }
        }
    }
}

// ---------------------------------------------------------------------------
// Per-step cell update + FC head + stage next step's x_t to f16.
// One block per batch row; thread j = hidden unit. Gates order: i,f,g,o.
// ---------------------------------------------------------------------------
__device__ __forceinline__ float lstm_sigmoid(float v) {
    return 1.0f / (1.0f + __expf(-v));
}
__device__ __forceinline__ float lstm_tanh(float v) {
    return 1.0f - 2.0f / (__expf(2.0f * v) + 1.0f);
}

__global__ __launch_bounds__(512)
void lstm_cell_step(const float* __restrict__ gates,   // [B, 4H]
                    float* __restrict__ c,             // [B, H] f32 state
                    _Float16* __restrict__ hf,         // [B, H] f16 state
                    _Float16* __restrict__ xh,         // [B, E] f16 stage (next x_t)
                    const float* __restrict__ x,       // [B, T, E] f32
                    const float* __restrict__ W_fc,    // [OUT, H]
                    const float* __restrict__ b_fc,    // [OUT]
                    float* __restrict__ out,           // [B, T, OUT]
                    int t) {
    const int b = blockIdx.x;       // 0..255
    const int j = threadIdx.x;      // 0..511
    const float* g = gates + (size_t)b * LSTM_G;

    const float ig = lstm_sigmoid(g[j]);
    const float fg = lstm_sigmoid(g[LSTM_H + j]);
    const float gg = lstm_tanh(g[2 * LSTM_H + j]);
    const float og = lstm_sigmoid(g[3 * LSTM_H + j]);

    const size_t sidx = (size_t)b * LSTM_H + j;
    const float cn = fg * c[sidx] + ig * gg;
    const float hn = og * lstm_tanh(cn);
    c[sidx]  = cn;
    hf[sidx] = (_Float16)hn;

    // Stage x[:, t+1, :] as f16 for the next step's GEMM (j spans E).
    if (t + 1 < LSTM_T) {
        xh[sidx] = (_Float16)x[((size_t)b * LSTM_T + (t + 1)) * LSTM_E + j];
    }

    // FC head: y = h @ W_fc^T + b_fc (OUT = 2), LDS tree reduction.
    __shared__ float s0[LSTM_H];
    __shared__ float s1[LSTM_H];
    s0[j] = hn * W_fc[j];
    s1[j] = hn * W_fc[LSTM_H + j];
    __syncthreads();
    for (int off = LSTM_H / 2; off > 0; off >>= 1) {
        if (j < off) { s0[j] += s0[j + off]; s1[j] += s1[j + off]; }
        __syncthreads();
    }
    if (j == 0) {
        float* yo = out + ((size_t)b * LSTM_T + t) * LSTM_OUT;
        yo[0] = s0[0] + b_fc[0];
        yo[1] = s1[0] + b_fc[1];
    }
}

// ---------------------------------------------------------------------------
// Host-side launcher. Workspace layout (~7 MB):
//   [0)        WihP  : 2048*512 f16  (2 MB)
//   [2 MB)     WhhP  : 2048*512 f16  (2 MB)
//   [4 MB)     hf    : 256*512  f16  (256 KB)
//   [4.25 MB)  xh    : 256*512  f16  (256 KB)
//   [4.5 MB)   c     : 256*512  f32  (512 KB)
//   [5 MB)     gates : 256*2048 f32  (2 MB)
// ---------------------------------------------------------------------------
extern "C" void kernel_launch(void* const* d_in, const int* in_sizes, int n_in,
                              void* d_out, int out_size, void* d_ws, size_t ws_size,
                              hipStream_t stream) {
    (void)in_sizes; (void)n_in; (void)out_size; (void)ws_size;

    const float* x    = (const float*)d_in[0];
    const float* W_ih = (const float*)d_in[1];
    const float* W_hh = (const float*)d_in[2];
    const float* b_ih = (const float*)d_in[3];
    const float* b_hh = (const float*)d_in[4];
    const float* W_fc = (const float*)d_in[5];
    const float* b_fc = (const float*)d_in[6];
    float* out = (float*)d_out;

    _Float16* WihP = (_Float16*)d_ws;
    _Float16* WhhP = WihP + (size_t)LSTM_G * LSTM_H;
    _Float16* hf   = WhhP + (size_t)LSTM_G * LSTM_H;
    _Float16* xh   = hf + (size_t)LSTM_B * LSTM_H;
    float* c       = (float*)(xh + (size_t)LSTM_B * LSTM_E);
    float* gates   = c + (size_t)LSTM_B * LSTM_H;

    // One-time (per call) weight repack + state/x0 init.
    lstm_pack_w<<<(LSTM_G * LSTM_H) / 256, 256, 0, stream>>>(W_ih, WihP);
    lstm_pack_w<<<(LSTM_G * LSTM_H) / 256, 256, 0, stream>>>(W_hh, WhhP);
    lstm_init_state<<<(LSTM_B * LSTM_H) / 256, 256, 0, stream>>>(c, hf, xh, x);

    // 512 sequential timesteps (recurrence on h).
    for (int t = 0; t < LSTM_T; ++t) {
        lstm_gemm_step<<<128, 128, 0, stream>>>(xh, hf, WihP, WhhP, b_ih, b_hh, gates);
        lstm_cell_step<<<LSTM_B, LSTM_H, 0, stream>>>(gates, c, hf, xh, x, W_fc, b_fc, out, t);
    }
}
